// VQGAN_11759620456817
// MI455X (gfx1250) — matrix operand order repaired and checked by reading, not compiled, not run on hardware
//
#include <hip/hip_runtime.h>

// ---------------------------------------------------------------------------
// Types for CDNA5 WMMA (wave32): bf16 A/B fragments (16 elems/lane = 8 VGPRs),
// f32 accumulator (8 elems/lane = 8 VGPRs).
// ---------------------------------------------------------------------------
typedef __attribute__((ext_vector_type(16))) __bf16 v16bf;
typedef __attribute__((ext_vector_type(8)))  __bf16 v8bf;
typedef __attribute__((ext_vector_type(8)))  float  v8f;
typedef __attribute__((ext_vector_type(4)))  float  v4f;

#define CONV_TM 32
#define CONV_TN 64
#define CONV_KS 64                 // K elements staged per barrier (2 WMMA k-steps)
#define LDS_PITCH (CONV_KS + 8)    // 72 bf16 = 144B rows: keeps 16B alignment

// ---------------------------------------------------------------------------
// Implicit-GEMM conv, NHWC / HWIO, bf16 WMMA with f32 accumulate.
// M = B*Ho*Wo output pixels, N = Cout, K = KH*KW*Cin iterated as
// nested (ky, kx, ci-chunk) loops -> no integer div/mod in the hot loop.
// Workgroup: 256 threads = 8 waves -> 32x64 output tile (2x4 of 16x16 WMMA).
// 64-wide K panel per sync -> 2 WMMAs per barrier pair.
// A panel in LDS row-major [M][K]; B panel transposed [N][K] so every
// fragment is two contiguous 16B LDS reads (ds_load_b128).
// Fast path (Cin % 64 == 0: every conv except conv_in): A staged with two
// predicated float4 loads; B staged with 16 grouped strided loads.
// Fused: optional bias, optional elementwise addsrc (residual), f32 out.
// ---------------------------------------------------------------------------
__global__ __launch_bounds__(256)
void conv2d_wmma_kernel(const float* __restrict__ in,
                        const float* __restrict__ wgt,
                        const float* __restrict__ bias,
                        const float* __restrict__ addsrc,
                        float* __restrict__ out,
                        int B, int H, int W, int Cin,
                        int KH, int KW, int stride, int pad,
                        int Ho, int Wo, int Cout)
{
  __shared__ __bf16 As[CONV_TM][LDS_PITCH];   // [pixel][k]
  __shared__ __bf16 Bt[CONV_TN][LDS_PITCH];   // [cout][k]  (transposed panel)

  const int tid  = threadIdx.x;
  const int lane = tid & 31;
  const int wave = tid >> 5;
  const int msub = wave >> 2;   // 0..1 : 16-row half of the 32-row tile
  const int nsub = wave & 3;    // 0..3 : 16-col slice of the 64-col tile

  const int P     = B * Ho * Wo;
  const int tileM = blockIdx.x * CONV_TM;
  const int tileN = blockIdx.y * CONV_TN;
  const bool fullk = ((Cin & (CONV_KS - 1)) == 0);   // uniform fast path

  // ---- A-panel staging: 32 pixels x 64 k / 256 thr = 8 each ---------------
  const int a_p  = tid >> 3;        // pixel row in tile (0..31)
  const int a_k0 = (tid & 7) * 8;   // 8 consecutive k (= ci) columns
  const int gp_a = tileM + a_p;
  int ab = 0, aoy = 0, aox = 0;
  const bool a_valid = (gp_a < P);
  if (a_valid) {
    ab = gp_a / (Ho * Wo);
    int r = gp_a - ab * (Ho * Wo);
    aoy = r / Wo;
    aox = r - aoy * Wo;
  }
  const int ayB = aoy * stride - pad;
  const int axB = aox * stride - pad;

  // ---- B-panel staging: 64 cols x 64 k / 256 thr = 16 each ----------------
  const int b_co = tid & 63;          // column (cout) within tile
  const int b_k0 = (tid >> 6) * 16;   // 16 consecutive k rows
  const int cog  = tileN + b_co;
  const bool co_ok = (cog < Cout);

  // ---- fragment addresses (loop-invariant) --------------------------------
  const int hl   = lane >> 4;                 // lane half
  const int mrow = msub * 16 + (lane & 15);   // A fragment row
  const int ncol = nsub * 16 + (lane & 15);   // B fragment column

  v8f acc = {0.f, 0.f, 0.f, 0.f, 0.f, 0.f, 0.f, 0.f};

  for (int ky = 0; ky < KH; ++ky) {
    const int iy = ayB + ky;
    const bool row_ok = a_valid & (iy >= 0) & (iy < H);
    for (int kx = 0; kx < KW; ++kx) {
      const int ix = axB + kx;
      const bool pix_ok = row_ok & (ix >= 0) & (ix < W);
      const float* inp = in + (((long long)ab * H + iy) * W + ix) * Cin;
      const long long wb = (long long)(ky * KW + kx) * Cin * Cout;

      for (int cc = 0; cc < Cin; cc += CONV_KS) {
        // ---- stage A panel (f32 -> bf16, zero-pad): one ds_store_b128 ----
        v8bf av;
        if (fullk) {
          v4f v0 = {0.f, 0.f, 0.f, 0.f};
          v4f v1 = {0.f, 0.f, 0.f, 0.f};
          if (pix_ok) {                       // 2x global_load_b128, grouped
            const float* ap = inp + cc + a_k0;
            v0 = *(const v4f*)ap;
            v1 = *(const v4f*)(ap + 4);
          }
#pragma unroll
          for (int j = 0; j < 4; ++j) { av[j] = (__bf16)v0[j]; av[4 + j] = (__bf16)v1[j]; }
        } else {
#pragma unroll
          for (int j = 0; j < 8; ++j) {
            const int ci = cc + a_k0 + j;
            const float v = (pix_ok && ci < Cin) ? inp[ci] : 0.f;
            av[j] = (__bf16)v;
          }
        }
        *(v8bf*)&As[a_p][a_k0] = av;

        // ---- stage B panel transposed: two ds_store_b128 ------------------
        const float* wp = wgt + wb + (long long)(cc + b_k0) * Cout + cog;
        v8bf bv0, bv1;
        if (fullk) {
          float wv[16];
#pragma unroll
          for (int j = 0; j < 16; ++j) wv[j] = 0.f;
          if (co_ok) {
#pragma unroll
            for (int j = 0; j < 16; ++j) wv[j] = wp[(long long)j * Cout];  // grouped loads
          }
#pragma unroll
          for (int j = 0; j < 8; ++j) { bv0[j] = (__bf16)wv[j]; bv1[j] = (__bf16)wv[8 + j]; }
          // prefetch next weight chunk (global_prefetch_b8; speculative-safe)
          __builtin_prefetch(wp + (long long)CONV_KS * Cout, 0, 3);
        } else {
#pragma unroll
          for (int j = 0; j < 8; ++j) {
            const int kl = b_k0 + j;
            const float v = (co_ok && (cc + kl) < Cin) ? wp[(long long)j * Cout] : 0.f;
            bv0[j] = (__bf16)v;
          }
#pragma unroll
          for (int j = 0; j < 8; ++j) {
            const int kl = b_k0 + 8 + j;
            const float v = (co_ok && (cc + kl) < Cin) ? wp[(long long)(8 + j) * Cout] : 0.f;
            bv1[j] = (__bf16)v;
          }
        }
        *(v8bf*)&Bt[b_co][b_k0]     = bv0;
        *(v8bf*)&Bt[b_co][b_k0 + 8] = bv1;

        __syncthreads();

        // ---- 2 WMMA k-steps per barrier: 8x ds_load_b128 per lane --------
#pragma unroll
        for (int s = 0; s < 2; ++s) {
          const int kb = 32 * s;
          const v8bf a_lo = *(const v8bf*)&As[mrow][kb + 8 * hl];
          const v8bf a_hi = *(const v8bf*)&As[mrow][kb + 16 + 8 * hl];
          const v8bf b_lo = *(const v8bf*)&Bt[ncol][kb + 8 * hl];
          const v8bf b_hi = *(const v8bf*)&Bt[ncol][kb + 16 + 8 * hl];
          const v16bf af = __builtin_shufflevector(a_lo, a_hi,
              0, 1, 2, 3, 4, 5, 6, 7, 8, 9, 10, 11, 12, 13, 14, 15);
          const v16bf bf = __builtin_shufflevector(b_lo, b_hi,
              0, 1, 2, 3, 4, 5, 6, 7, 8, 9, 10, 11, 12, 13, 14, 15);
          acc = __builtin_amdgcn_wmma_f32_16x16x32_bf16(
              /*neg_a=*/false, af, /*neg_b=*/false, bf,
              /*c_mod=*/(short)0, acc, /*reuse_a=*/false, /*reuse_b=*/false);
        }
        __syncthreads();
      }
    }
  }

  // ---- epilogue: C layout -> lane = N col, VGPR r = M row r + 8*(lane>=16)
  const int n_g = tileN + nsub * 16 + (lane & 15);
  float bvs = 0.f;
  if (bias && n_g < Cout) bvs = bias[n_g];
#pragma unroll
  for (int r = 0; r < 8; ++r) {
    const int m  = msub * 16 + (lane >> 4) * 8 + r;
    const int gp = tileM + m;
    if (gp < P && n_g < Cout) {
      const long long o = (long long)gp * Cout + n_g;
      float v = acc[r] + bvs;
      if (addsrc) v += addsrc[o];
      out[o] = v;
    }
  }
}

// ---------------------------------------------------------------------------
// GroupNorm (8 groups) + SiLU, NHWC. One block per (batch, group).
// ---------------------------------------------------------------------------
__global__ __launch_bounds__(256)
void gn_silu_kernel(const float* __restrict__ in,
                    const float* __restrict__ gs,
                    const float* __restrict__ gb,
                    float* __restrict__ out,
                    int HW, int C, int Cg)
{
  const int b   = blockIdx.x >> 3;
  const int g   = blockIdx.x & 7;
  const int tid = threadIdx.x;
  const long long base = (long long)b * HW * C;
  const int cnt = HW * Cg;

  float sum = 0.f, sq = 0.f;
  for (int t = tid; t < cnt; t += 256) {
    const int pix = t / Cg;
    const int c   = g * Cg + (t - pix * Cg);
    const float v = in[base + (long long)pix * C + c];
    sum += v; sq += v * v;
  }
  __shared__ float rs[256], rq[256];
  rs[tid] = sum; rq[tid] = sq;
  __syncthreads();
  for (int off = 128; off > 0; off >>= 1) {
    if (tid < off) { rs[tid] += rs[tid + off]; rq[tid] += rq[tid + off]; }
    __syncthreads();
  }
  __shared__ float s_mean, s_rstd;
  if (tid == 0) {
    const float mean = rs[0] / (float)cnt;
    const float var  = rq[0] / (float)cnt - mean * mean;
    s_mean = mean;
    s_rstd = rsqrtf(var + 1e-5f);
  }
  __syncthreads();
  const float mean = s_mean, rstd = s_rstd;
  for (int t = tid; t < cnt; t += 256) {
    const int pix = t / Cg;
    const int c   = g * Cg + (t - pix * Cg);
    const long long o = base + (long long)pix * C + c;
    const float y = (in[o] - mean) * rstd * gs[c] + gb[c];
    out[o] = y / (1.f + __expf(-y));  // SiLU
  }
}

// ---------------------------------------------------------------------------
// Nearest-neighbor 2x upsample, NHWC, grid-stride.
// ---------------------------------------------------------------------------
__global__ __launch_bounds__(256)
void upsample2x_kernel(const float* __restrict__ in, float* __restrict__ out,
                       int B, int H, int W, int C, long long n)
{
  for (long long i = (long long)blockIdx.x * 256 + threadIdx.x; i < n;
       i += (long long)gridDim.x * 256) {
    const int c = (int)(i % C);
    long long t = i / C;
    const int x = (int)(t % (2 * W)); t /= (2 * W);
    const int y = (int)(t % (2 * H));
    const int b = (int)(t / (2 * H));
    out[i] = in[(((long long)b * H + (y >> 1)) * W + (x >> 1)) * C + c];
  }
}

// ---------------------------------------------------------------------------
// VQ: per-token argmin over codebook, gather q, per-token squared error.
// One block per token (D=256, K=1024). z_q = q (straight-through forward).
// ---------------------------------------------------------------------------
__global__ __launch_bounds__(256)
void vq_kernel(const float* __restrict__ z, const float* __restrict__ cb,
               float* __restrict__ zq, float* __restrict__ idxf,
               float* __restrict__ err, int D, int K)
{
  const int t   = blockIdx.x;
  const int tid = threadIdx.x;
  __shared__ float zl[256];
  for (int d = tid; d < D; d += 256) zl[d] = z[(long long)t * D + d];
  __syncthreads();

  float best = 3.4e38f;
  int   bi   = 0x7fffffff;
  for (int k = tid; k < K; k += 256) {
    const float* c = cb + (long long)k * D;
    float dist = 0.f;
    for (int d = 0; d < D; ++d) { const float df = zl[d] - c[d]; dist += df * df; }
    if (dist < best) { best = dist; bi = k; }
  }
  __shared__ float bd[256];
  __shared__ int   bx[256];
  bd[tid] = best; bx[tid] = bi;
  __syncthreads();
  for (int off = 128; off > 0; off >>= 1) {
    if (tid < off) {
      if (bd[tid + off] < bd[tid] ||
          (bd[tid + off] == bd[tid] && bx[tid + off] < bx[tid])) {
        bd[tid] = bd[tid + off]; bx[tid] = bx[tid + off];
      }
    }
    __syncthreads();
  }
  const int kb = bx[0];

  float es = 0.f;
  for (int d = tid; d < D; d += 256) {
    const float q = cb[(long long)kb * D + d];
    zq[(long long)t * D + d] = q;
    const float df = q - zl[d];
    es += df * df;
  }
  __syncthreads();
  bd[tid] = es;
  __syncthreads();
  for (int off = 128; off > 0; off >>= 1) {
    if (tid < off) bd[tid] += bd[tid + off];
    __syncthreads();
  }
  if (tid == 0) { err[t] = bd[0]; idxf[t] = (float)kb; }
}

__global__ __launch_bounds__(256)
void vq_finalize_kernel(const float* __restrict__ err, float* __restrict__ closs,
                        float* __restrict__ eloss, int T, float invTD)
{
  __shared__ float rs[256];
  float s = 0.f;
  for (int t = threadIdx.x; t < T; t += 256) s += err[t];
  rs[threadIdx.x] = s;
  __syncthreads();
  for (int off = 128; off > 0; off >>= 1) {
    if (threadIdx.x < off) rs[threadIdx.x] += rs[threadIdx.x + off];
    __syncthreads();
  }
  if (threadIdx.x == 0) {
    const float mse = rs[0] * invTD;
    *closs = 0.25f * mse;  // BETA * mean((q - z_e)^2)
    *eloss = mse;          // mean((q - z_e)^2)
  }
}

// ---------------------------------------------------------------------------
// Host-side orchestration
// ---------------------------------------------------------------------------
struct ConvP { const float* w; const float* b; int kh, kw, cin, cout; };
struct GNP   { const float* s; const float* b; };
struct ResP  { ConvP c1, c2; GNP g1, g2; const float* proj; int cin, cout; };

static void launch_conv(hipStream_t st, const float* in, int B, int H, int W,
                        const ConvP& c, int stride, int pad,
                        const float* addsrc, float* out, int& Ho, int& Wo)
{
  Ho = (H + stride - 1) / stride;   // JAX 'SAME'
  Wo = (W + stride - 1) / stride;
  const int P = B * Ho * Wo;
  dim3 grid((P + CONV_TM - 1) / CONV_TM, (c.cout + CONV_TN - 1) / CONV_TN);
  conv2d_wmma_kernel<<<grid, 256, 0, st>>>(in, c.w, c.b, addsrc, out,
                                           B, H, W, c.cin, c.kh, c.kw,
                                           stride, pad, Ho, Wo, c.cout);
}

static void launch_gnsilu(hipStream_t st, const float* in, int B, int H, int W,
                          int C, const GNP& g, float* out)
{
  gn_silu_kernel<<<B * 8, 256, 0, st>>>(in, g.s, g.b, out, H * W, C, C / 8);
}

// out lands in sC. x used as fused residual only when no proj.
static void res_exec(hipStream_t st, const ResP& r, const float* x,
                     int B, int H, int W, float* sA, float* sB, float* sC)
{
  int ho, wo;
  launch_gnsilu(st, x, B, H, W, r.cin, r.g1, sA);                     // t1
  launch_conv(st, sA, B, H, W, r.c1, 1, 1, nullptr, sB, ho, wo);      // t2
  launch_gnsilu(st, sB, B, H, W, r.cout, r.g2, sA);                   // t3
  if (!r.proj) {
    launch_conv(st, sA, B, H, W, r.c2, 1, 1, x, sC, ho, wo);          // h4 + x
  } else {
    launch_conv(st, sA, B, H, W, r.c2, 1, 1, nullptr, sB, ho, wo);    // h4
    ConvP pj; pj.w = r.proj; pj.b = nullptr;
    pj.kh = 1; pj.kw = 1; pj.cin = r.cout; pj.cout = r.cout;
    launch_conv(st, sB, B, H, W, pj, 1, 0, sB, sC, ho, wo);           // proj(h4)+h4
  }
}

extern "C" void kernel_launch(void* const* d_in, const int* in_sizes, int n_in,
                              void* d_out, int out_size, void* d_ws, size_t ws_size,
                              hipStream_t stream)
{
  (void)in_sizes; (void)n_in; (void)out_size; (void)ws_size;

  // ---- pytree flatten order: x, codebook, decoder, encoder (dict keys sorted)
  const float* x_in     = (const float*)d_in[0];
  const float* codebook = (const float*)d_in[1];
  int ip = 2;
  auto P = [&]() -> const float* { return (const float*)d_in[ip++]; };

  auto parse_res = [&](int cin, int cout) -> ResP {
    ResP r; r.cin = cin; r.cout = cout;
    r.c1.b = P(); r.c1.w = P(); r.c1.kh = 3; r.c1.kw = 3; r.c1.cin = cin;  r.c1.cout = cout;
    r.c2.w = P(); r.c2.b = nullptr; r.c2.kh = 3; r.c2.kw = 3; r.c2.cin = cout; r.c2.cout = cout;
    r.g1.b = P(); r.g1.s = P();
    r.g2.b = P(); r.g2.s = P();
    r.proj = (cin != cout) ? P() : nullptr;
    return r;
  };

  // ---- decoder params (keys: conv_in, conv_out, gn, mid, stages) ----
  ConvP d_ci; d_ci.b = P(); d_ci.w = P(); d_ci.kh = 3; d_ci.kw = 3; d_ci.cin = 256; d_ci.cout = 512;
  ConvP d_co; d_co.b = P(); d_co.w = P(); d_co.kh = 3; d_co.kw = 3; d_co.cin = 128; d_co.cout = 3;
  GNP   d_gn; d_gn.b = P(); d_gn.s = P();
  ResP  d_mid[2] = { parse_res(512, 512), parse_res(512, 512) };
  ResP  d_res[4][2];
  ConvP d_up[4]; bool d_hasup[4];
  const int d_cfg[4][2] = { {512, 512}, {512, 256}, {256, 256}, {256, 128} };
  for (int s = 0; s < 4; ++s) {
    const int f = d_cfg[s][1];
    d_res[s][0] = parse_res(d_cfg[s][0], f);   // stage keys: 'res' then 'up'
    d_res[s][1] = parse_res(f, f);
    d_hasup[s] = (s < 3);
    if (d_hasup[s]) {
      d_up[s].b = P(); d_up[s].w = P();
      d_up[s].kh = 3; d_up[s].kw = 3; d_up[s].cin = f; d_up[s].cout = f;
    }
  }

  // ---- encoder params (keys: conv_in, conv_out, gn, mid, stages) ----
  ConvP e_ci; e_ci.w = P(); e_ci.b = nullptr; e_ci.kh = 3; e_ci.kw = 3; e_ci.cin = 3;   e_ci.cout = 128;
  ConvP e_co; e_co.b = P(); e_co.w = P();     e_co.kh = 1; e_co.kw = 1; e_co.cin = 512; e_co.cout = 256;
  GNP   e_gn; e_gn.b = P(); e_gn.s = P();
  ResP  e_mid[2] = { parse_res(512, 512), parse_res(512, 512) };
  ResP  e_res[4][2];
  ConvP e_dn[4]; bool e_hasdn[4];
  const int e_cin[4] = {128, 128, 256, 256};
  const int e_f[4]   = {128, 256, 256, 512};
  for (int s = 0; s < 4; ++s) {
    e_hasdn[s] = (s < 3);
    if (e_hasdn[s]) {                          // stage keys: 'down' then 'res'
      e_dn[s].b = P(); e_dn[s].w = P();
      e_dn[s].kh = 4; e_dn[s].kw = 4; e_dn[s].cin = e_f[s]; e_dn[s].cout = e_f[s];
    }
    e_res[s][0] = parse_res(e_cin[s], e_f[s]);
    e_res[s][1] = parse_res(e_f[s], e_f[s]);
  }

  // ---- workspace slabs (4 x 64MB f32) + err buffer ----
  const long long SLAB = 16777216LL;   // 4*128*128*256 floats
  float* base = (float*)d_ws;
  float* slab[4];
  for (int i = 0; i < 4; ++i) slab[i] = base + (long long)i * SLAB;
  float* errbuf = base + 4 * SLAB;

  int Bn = 4, H = 128, W = 128, C = 0, xi = 0, ho, wo;
  auto oth = [&](int k) -> int {
    int c = 0;
    for (int i = 0; i < 4; ++i) { if (i != xi) { if (c == k) return i; ++c; } }
    return 0;
  };
  auto run_res = [&](const ResP& r) {
    const int a = oth(0), b = oth(1), cc = oth(2);
    res_exec(stream, r, slab[xi], Bn, H, W, slab[a], slab[b], slab[cc]);
    xi = cc; C = r.cout;
  };

  // ================= encoder =================
  launch_conv(stream, x_in, Bn, H, W, e_ci, 1, 1, nullptr, slab[0], ho, wo);
  xi = 0; C = 128;
  for (int s = 0; s < 4; ++s) {
    run_res(e_res[s][0]);
    run_res(e_res[s][1]);
    if (e_hasdn[s]) {
      const int o = oth(0);
      launch_conv(stream, slab[xi], Bn, H, W, e_dn[s], 2, 1, nullptr, slab[o], ho, wo);
      xi = o; H = ho; W = wo;
    }
  }
  run_res(e_mid[0]); run_res(e_mid[1]);
  { const int o = oth(0); launch_gnsilu(stream, slab[xi], Bn, H, W, C, e_gn, slab[o]); xi = o; }
  { const int o = oth(0); launch_conv(stream, slab[xi], Bn, H, W, e_co, 1, 0, nullptr, slab[o], ho, wo); xi = o; C = 256; }
  // z_e in slab[xi]: [4,16,16,256]

  // ================= quantize =================
  float* out_f = (float*)d_out;
  const long long OFF_ZQ = 196608, OFF_CL = 458752, OFF_EL = 458753, OFF_IDX = 458754;
  float* zq = out_f + OFF_ZQ;
  vq_kernel<<<1024, 256, 0, stream>>>(slab[xi], codebook, zq, out_f + OFF_IDX,
                                      errbuf, 256, 1024);
  vq_finalize_kernel<<<1, 256, 0, stream>>>(errbuf, out_f + OFF_CL, out_f + OFF_EL,
                                            1024, 1.f / (1024.f * 256.f));

  // ================= decoder =================
  H = 16; W = 16;
  launch_conv(stream, zq, Bn, H, W, d_ci, 1, 1, nullptr, slab[0], ho, wo);
  xi = 0; C = 512;
  run_res(d_mid[0]); run_res(d_mid[1]);
  for (int s = 0; s < 4; ++s) {
    run_res(d_res[s][0]);
    run_res(d_res[s][1]);
    if (d_hasup[s]) {
      const int o0 = oth(0), o1 = oth(1);
      const long long n = (long long)Bn * (2 * H) * (2 * W) * C;
      upsample2x_kernel<<<4096, 256, 0, stream>>>(slab[xi], slab[o0], Bn, H, W, C, n);
      H *= 2; W *= 2;
      launch_conv(stream, slab[o0], Bn, H, W, d_up[s], 1, 1, nullptr, slab[o1], ho, wo);
      xi = o1;
    }
  }
  { const int o = oth(0); launch_gnsilu(stream, slab[xi], Bn, H, W, C, d_gn, slab[o]); xi = o; }
  launch_conv(stream, slab[xi], Bn, H, W, d_co, 1, 1, nullptr, out_f, ho, wo);  // x_recon
}
